// GCNLayer_91173565759712
// MI455X (gfx1250) — matrix operand order repaired and checked by reading, not compiled, not run on hardware
//
#include <hip/hip_runtime.h>
#include <cstdint>
#include <cstddef>

// ---------------------------------------------------------------------------
// GCN layer: out = relu( (D A_hat D x) W^T ),  A_hat = I | edges (set semantics)
// N=8192 nodes, E=262144 edges, D_IN=D_OUT=256.
// Strategy: bitmask adjacency (exact dedupe) -> popcount degrees -> sparse
// aggregation in fp32 -> dense GEMM+ReLU on v_wmma_f32_16x16x32_f16.
// ---------------------------------------------------------------------------

#define GCN_N     8192
#define GCN_E     262144
#define GCN_D     256
#define GCN_WORDS 256          // 8192 bits / 32 per adjacency row
#define GCN_EPS   1e-5f

typedef __attribute__((ext_vector_type(16))) _Float16 v16h;
typedef __attribute__((ext_vector_type(8)))  _Float16 v8h;
typedef __attribute__((ext_vector_type(8)))  float    v8f;

// --- 1) zero the bitmask and set the diagonal (identity part of A_hat) ------
__global__ void gcn_init_mask(uint32_t* __restrict__ mask) {
    int w = blockIdx.x * blockDim.x + threadIdx.x;      // word index, 2M words
    int row  = w >> 8;                                  // node id
    int cw   = w & (GCN_WORDS - 1);                     // word within row
    uint32_t v = (cw == (row >> 5)) ? (1u << (row & 31)) : 0u;
    mask[w] = v;
}

// --- 2) scatter edges: A_hat[src, dst] = 1 (atomicOr dedupes for free) ------
__global__ void gcn_scatter_edges(const long long* __restrict__ ei,
                                  uint32_t* __restrict__ mask) {
    int e = blockIdx.x * blockDim.x + threadIdx.x;
    int src = (int)ei[e];
    int dst = (int)ei[GCN_E + e];
    atomicOr(&mask[(size_t)src * GCN_WORDS + (dst >> 5)], 1u << (dst & 31));
}

// --- 3) degrees via popcount, d_inv = rsqrt(deg + eps); one wave per node ---
__global__ void gcn_deg_inv(const uint32_t* __restrict__ mask,
                            float* __restrict__ dinv) {
    int wave = threadIdx.x >> 5;                        // 8 waves / block
    int lane = threadIdx.x & 31;
    int node = blockIdx.x * 8 + wave;
    const uint32_t* row = mask + (size_t)node * GCN_WORDS;
    int s = 0;
#pragma unroll
    for (int t = 0; t < GCN_WORDS / 32; ++t)            // coalesced across lanes
        s += __popc(row[lane + 32 * t]);
#pragma unroll
    for (int off = 16; off > 0; off >>= 1)
        s += __shfl_xor(s, off, 32);
    if (lane == 0)
        dinv[node] = rsqrtf((float)s + GCN_EPS);
}

// --- 4) W (fp32) -> f16 once -----------------------------------------------
__global__ void gcn_conv_w(const float* __restrict__ W, _Float16* __restrict__ wh) {
    int i = blockIdx.x * blockDim.x + threadIdx.x;      // 65536 elements
    wh[i] = (_Float16)W[i];
}

// --- 5) sparse aggregation: g[i,:] = dinv[i] * sum_{j in N(i)} dinv[j]*x[j,:]
// One block (256 threads) per node; thread t owns feature t. Bit-scan the
// row's bitmask (uniform across the block), gather x rows coalesced.
__global__ void gcn_aggregate(const uint32_t* __restrict__ mask,
                              const float* __restrict__ dinv,
                              const float* __restrict__ x,
                              _Float16* __restrict__ gh) {
    int i = blockIdx.x;
    int t = threadIdx.x;
    const uint32_t* row = mask + (size_t)i * GCN_WORDS;
    float acc = 0.0f;
    for (int w = 0; w < GCN_WORDS; ++w) {
        uint32_t word = row[w];                         // same addr: broadcast
        while (word) {
            int b = __builtin_ctz(word);
            word &= word - 1;
            int j = (w << 5) + b;
            acc += dinv[j] * x[(size_t)j * GCN_D + t];  // coalesced over t
        }
    }
    gh[(size_t)i * GCN_D + t] = (_Float16)(dinv[i] * acc);
}

// --- 6) out = relu(g @ W^T) on WMMA f16->f32 --------------------------------
// Each wave owns a 16x16 output tile; K=256 -> 8 x v_wmma_f32_16x16x32_f16.
// Layouts per CDNA5 ISA 7.12.2 (wave32).
__global__ void gcn_gemm_relu_wmma(const _Float16* __restrict__ gh,
                                   const _Float16* __restrict__ wh,
                                   float* __restrict__ out) {
    int wave   = threadIdx.x >> 5;                      // 8 waves / block
    int lane   = threadIdx.x & 31;
    int tileId = blockIdx.x * 8 + wave;                 // 512 * 16 tiles
    int mTile  = tileId >> 4;                           // 0..511
    int nTile  = tileId & 15;                           // 0..15
    int h      = lane >> 4;                             // K-half selector
    int mn     = lane & 15;                             // A row / B column

    const _Float16* arow = gh + (size_t)(mTile * 16 + mn) * GCN_D;
    const _Float16* brow = wh + (size_t)(nTile * 16 + mn) * GCN_D; // W[n, :]

    v8f c = {};
#pragma unroll
    for (int kk = 0; kk < GCN_D; kk += 32) {
        // A 16x32: lane m=mn; VGPR0-3 <- K = kk+8h .. +7 ; VGPR4-7 <- K = kk+16+8h .. +7
        v8h a0 = *(const v8h*)(arow + kk + 8 * h);
        v8h a1 = *(const v8h*)(arow + kk + 16 + 8 * h);
        // B 32x16: lane n=mn; VGPR0-3 <- K = kk+8h .. +7 ; VGPR4-7 <- K = kk+16+8h .. +7
        v8h b0 = *(const v8h*)(brow + kk + 8 * h);
        v8h b1 = *(const v8h*)(brow + kk + 16 + 8 * h);
        v16h a, b;
#pragma unroll
        for (int q = 0; q < 8; ++q) {
            a[q] = a0[q]; a[q + 8] = a1[q];
            b[q] = b0[q]; b[q + 8] = b1[q];
        }
        // (neg_a, A, neg_b, B, c_mod, C, reuse_a, reuse_b)
        c = __builtin_amdgcn_wmma_f32_16x16x32_f16(false, a, false, b,
                                                   (short)0, c, false, false);
    }
    // C/D: VGPR r holds (row = 8h + r, col = mn); fuse ReLU into epilogue.
    int n = nTile * 16 + mn;
#pragma unroll
    for (int r = 0; r < 8; ++r) {
        int m = mTile * 16 + h * 8 + r;
        out[(size_t)m * GCN_D + n] = fmaxf(c[r], 0.0f);
    }
}

// ---------------------------------------------------------------------------
extern "C" void kernel_launch(void* const* d_in, const int* in_sizes, int n_in,
                              void* d_out, int out_size, void* d_ws, size_t ws_size,
                              hipStream_t stream) {
    const float*     x  = (const float*)d_in[0];
    const long long* ei = (const long long*)d_in[1];   // int64 [2, E] flat
    const float*     W  = (const float*)d_in[2];
    float* out = (float*)d_out;

    // workspace: mask 8MB | dinv 32KB | g_f16 4MB | W_f16 128KB  (~12.2MB)
    char* ws = (char*)d_ws;
    uint32_t*  mask = (uint32_t*)ws;
    float*     dinv = (float*)(ws + (size_t)8 * 1024 * 1024);
    _Float16*  gh   = (_Float16*)(ws + (size_t)8 * 1024 * 1024 + 32 * 1024);
    _Float16*  wh   = (_Float16*)(ws + (size_t)8 * 1024 * 1024 + 32 * 1024
                                      + (size_t)4 * 1024 * 1024);

    gcn_init_mask     <<<(GCN_N * GCN_WORDS) / 256, 256, 0, stream>>>(mask);
    gcn_scatter_edges <<<GCN_E / 256,               256, 0, stream>>>(ei, mask);
    gcn_conv_w        <<<(GCN_D * GCN_D) / 256,     256, 0, stream>>>(W, wh);
    gcn_deg_inv       <<<GCN_N / 8,                 256, 0, stream>>>(mask, dinv);
    gcn_aggregate     <<<GCN_N,                     256, 0, stream>>>(mask, dinv, x, gh);
    gcn_gemm_relu_wmma<<<(GCN_N / 16) * (GCN_D / 16) / 8, 256, 0, stream>>>(gh, wh, out);
}
